// MultiHeadAttention_39676907887747
// MI455X (gfx1250) — matrix operand order repaired and checked by reading, not compiled, not run on hardware
//
#include <hip/hip_runtime.h>

// ---------------------------------------------------------------------------
// MI455X (gfx1250) fused multi-head ReLU-attention, bf16 WMMA pipeline.
//   B=2, S=2048, D=1024, H=16, dk=64.
// All matmuls use v_wmma_f32_16x16x32_bf16 (wave32, 16x16 tiles, K=32).
// 32-row M-tiles per wave: every B fragment feeds two WMMAs (~21 FLOP/B of
// operand traffic vs ~13 at 16 rows) since the matrix pipe, not HBM, is the
// limiter once the fused attention keeps the 134M-element score matrix out of
// global memory (it lives in registers + a 2.5KB/wave LDS staging tile).
// ---------------------------------------------------------------------------

typedef __bf16 bf16;
typedef __attribute__((ext_vector_type(16))) __bf16 v16bf;
typedef __attribute__((ext_vector_type(8)))  __bf16 v8bf;
typedef __attribute__((ext_vector_type(8)))  float   v8f;

#define S_LEN  2048
#define DMODEL 1024
#define NHEAD  16
#define DK     64
#define NBATCH 2

__device__ __forceinline__ v8f wmma_bf16(v16bf a, v16bf b, v8f c) {
  return __builtin_amdgcn_wmma_f32_16x16x32_bf16(
      /*neg_a=*/false, a, /*neg_b=*/false, b,
      /*c_mod=*/(short)0, c, /*reuse_a=*/false, /*reuse_b=*/false);
}

// A fragment (16x32 bf16, row-major source, stride lda elements).
// ISA 7.12.2: lane L holds row m=L%16; 16 elems = two contiguous 8-elem
// chunks at k = 8*(L/16) and k = 16 + 8*(L/16).
__device__ __forceinline__ v16bf load_a_frag(const bf16* __restrict__ base, int lda) {
  int lane = threadIdx.x & 31;
  int m = lane & 15, hf = lane >> 4;
  const bf16* p = base + (size_t)m * lda + hf * 8;
  v8bf lo = *(const v8bf*)(p);
  v8bf hi = *(const v8bf*)(p + 16);
  v16bf a;
#pragma unroll
  for (int i = 0; i < 8; ++i) { a[i] = lo[i]; a[i + 8] = hi[i]; }
  return a;
}

// B fragment (32x16 bf16, [K,N] row-major source, stride ldb elements).
// lane L holds column n=L%16; 16 elems at k = 8*(L/16)+i and k = 16 + 8*(L/16)+i.
__device__ __forceinline__ v16bf load_b_frag(const bf16* __restrict__ base, int ldb) {
  int lane = threadIdx.x & 31;
  int n = lane & 15, hf = lane >> 4;
  const bf16* p = base + (size_t)(hf * 8) * ldb + n;
  v16bf b;
#pragma unroll
  for (int i = 0; i < 8; ++i) { b[i] = p[(size_t)i * ldb]; b[i + 8] = p[(size_t)(16 + i) * ldb]; }
  return b;
}

// ---------------------------------------------------------------------------
__global__ void cvt_f32_bf16(const float* __restrict__ in, bf16* __restrict__ out, int n) {
  int i = blockIdx.x * blockDim.x + threadIdx.x;
  if (i < n) out[i] = (bf16)in[i];
}

// W is [N,K] row-major fp32; produce Wt [K,N] bf16 so B-fragments are contiguous.
__global__ void transpose_cvt(const float* __restrict__ in, bf16* __restrict__ out) {
  int i = blockIdx.x * blockDim.x + threadIdx.x;   // 0 .. 1M-1
  int k = i & (DMODEL - 1);
  int n = i >> 10;
  out[(size_t)k * DMODEL + n] = (bf16)in[(size_t)n * DMODEL + k];
}

// ---------------------------------------------------------------------------
// QKV projections: y = x @ W^T.  One wave => one 32x64 output tile
// (two 16-row subtiles sharing every B fragment).
// q,v stored [B,H,S,dk]; k stored transposed [B,H,dk,S] for the q.k^T stage.
__global__ void qkv_gemm(const bf16* __restrict__ xb,
                         const bf16* __restrict__ wqt, const bf16* __restrict__ wkt,
                         const bf16* __restrict__ wvt,
                         bf16* __restrict__ qb, bf16* __restrict__ ktb, bf16* __restrict__ vb) {
  int wid   = blockIdx.x * (blockDim.x >> 5) + (threadIdx.x >> 5);
  int which = wid >> 11;          // 0=q 1=k 2=v   (2048 tiles each)
  int rem   = wid & 2047;
  int mtile = rem >> 4;           // 0..127 (32-row tiles over B*S)
  int ntile = rem & 15;           // 0..15 == head index (64-wide n tile)
  const bf16* wt = (which == 0) ? wqt : (which == 1) ? wkt : wvt;
  const bf16* A  = xb + (size_t)mtile * 32 * DMODEL;
  const bf16* Bt = wt + ntile * 64;

  v8f acc[2][4] = {};
  for (int k0 = 0; k0 < DMODEL; k0 += 32) {
    v16bf af0 = load_a_frag(A + k0, DMODEL);
    v16bf af1 = load_a_frag(A + (size_t)16 * DMODEL + k0, DMODEL);
#pragma unroll
    for (int j = 0; j < 4; ++j) {
      v16bf bfr = load_b_frag(Bt + (size_t)k0 * DMODEL + j * 16, DMODEL);
      acc[0][j] = wmma_bf16(af0, bfr, acc[0][j]);
      acc[1][j] = wmma_bf16(af1, bfr, acc[1][j]);
    }
  }

  int lane = threadIdx.x & 31, nn = lane & 15, hf = lane >> 4;
  if (which != 1) {
    bf16* dst = (which == 0) ? qb : vb;     // [B,H,S,dk]
#pragma unroll
    for (int t = 0; t < 2; ++t)
#pragma unroll
      for (int j = 0; j < 4; ++j)
#pragma unroll
        for (int r = 0; r < 8; ++r) {
          int mg = mtile * 32 + t * 16 + r + 8 * hf;
          int b = mg >> 11, si = mg & (S_LEN - 1);
          dst[(((size_t)(b * NHEAD + ntile) * S_LEN + si) * DK) + j * 16 + nn] = (bf16)acc[t][j][r];
        }
  } else {                                   // k transposed: [B,H,dk,S]
#pragma unroll
    for (int t = 0; t < 2; ++t)
#pragma unroll
      for (int j = 0; j < 4; ++j)
#pragma unroll
        for (int r = 0; r < 8; ++r) {
          int mg = mtile * 32 + t * 16 + r + 8 * hf;
          int b = mg >> 11, si = mg & (S_LEN - 1);
          int d = j * 16 + nn;
          ktb[((size_t)(b * NHEAD + ntile) * DK + d) * S_LEN + si] = (bf16)acc[t][j][r];
        }
  }
}

// ---------------------------------------------------------------------------
// Fused attention: out = relu(q k^T * scale + mask) @ v, per (b,h).
// One wave owns 32 query rows (2 subtiles); loops over S in 32-wide key
// blocks. k-score and v fragments are each reused by both subtiles.
__global__ void attn_relu(const bf16* __restrict__ qb, const bf16* __restrict__ ktb,
                          const bf16* __restrict__ vb, const float* __restrict__ mask,
                          bf16* __restrict__ ob) {
  __shared__ bf16 smem[8 * 32 * 40];   // 8 waves * (32 rows x 40 padded cols)
  int wv    = threadIdx.x >> 5;
  int sbase = wv * 32 * 40;
  int wid   = blockIdx.x * 8 + wv;
  int bh    = wid >> 6;                // 0..31
  int mtile = wid & 63;                // 0..63 (32 query rows each)
  int lane  = threadIdx.x & 31, nn = lane & 15, hf = lane >> 4;

  const bf16* qA = qb  + (size_t)bh * S_LEN * DK + (size_t)mtile * 32 * DK;
  const bf16* kt = ktb + (size_t)bh * DK * S_LEN;
  const bf16* vv = vb  + (size_t)bh * S_LEN * DK;

  v16bf aq[2][2];
#pragma unroll
  for (int t = 0; t < 2; ++t) {
    aq[t][0] = load_a_frag(qA + (size_t)t * 16 * DK,      DK);  // d = 0..31
    aq[t][1] = load_a_frag(qA + (size_t)t * 16 * DK + 32, DK);  // d = 32..63
  }
  v8f oacc[2][4] = {};
  const float scale = 0.125f;          // 1/sqrt(64)

  for (int n0 = 0; n0 < S_LEN; n0 += 32) {
    // ---- score B fragments (shared by both query subtiles) ---------------
    v16bf bk0a = load_b_frag(kt + n0,                   S_LEN);
    v16bf bk0b = load_b_frag(kt + 32 * S_LEN + n0,      S_LEN);
    v16bf bk1a = load_b_frag(kt + n0 + 16,              S_LEN);
    v16bf bk1b = load_b_frag(kt + 32 * S_LEN + n0 + 16, S_LEN);

#pragma unroll
    for (int t = 0; t < 2; ++t) {
      v8f s0 = {}, s1 = {};
      s0 = wmma_bf16(aq[t][0], bk0a, s0);
      s0 = wmma_bf16(aq[t][1], bk0b, s0);
      s1 = wmma_bf16(aq[t][0], bk1a, s1);
      s1 = wmma_bf16(aq[t][1], bk1b, s1);

      // ---- scale + mask + relu, stage to LDS in row-major ----------------
#pragma unroll
      for (int r = 0; r < 8; ++r) {
        int m  = t * 16 + r + 8 * hf;
        int mq = mtile * 32 + m;
        float v0 = fmaxf(s0[r] * scale + mask[(size_t)mq * S_LEN + n0 + nn],      0.0f);
        float v1 = fmaxf(s1[r] * scale + mask[(size_t)mq * S_LEN + n0 + 16 + nn], 0.0f);
        smem[sbase + m * 40 + nn]      = (bf16)v0;
        smem[sbase + m * 40 + 16 + nn] = (bf16)v1;
      }
    }
    asm volatile("s_wait_dscnt 0x0" ::: "memory");  // LDS RAW fence (same wave, in-order DS pipe)

    // ---- reload scores as A fragments ------------------------------------
    v16bf as[2];
#pragma unroll
    for (int t = 0; t < 2; ++t) {
      const bf16* p = &smem[sbase + (t * 16 + (lane & 15)) * 40 + hf * 8];
      v8bf lo = *(const v8bf*)(p);
      v8bf hi = *(const v8bf*)(p + 16);
#pragma unroll
      for (int i = 0; i < 8; ++i) { as[t][i] = lo[i]; as[t][i + 8] = hi[i]; }
    }

    // ---- multiply by v (each v fragment reused by both subtiles) ---------
#pragma unroll
    for (int j = 0; j < 4; ++j) {
      v16bf bv = load_b_frag(vv + (size_t)n0 * DK + j * 16, DK);
      oacc[0][j] = wmma_bf16(as[0], bv, oacc[0][j]);
      oacc[1][j] = wmma_bf16(as[1], bv, oacc[1][j]);
    }
  }

  // ---- store merged-head bf16 output [B,S,D] -----------------------------
  int b = bh >> 4, h = bh & 15;
#pragma unroll
  for (int t = 0; t < 2; ++t)
#pragma unroll
    for (int j = 0; j < 4; ++j)
#pragma unroll
      for (int r = 0; r < 8; ++r) {
        int si = mtile * 32 + t * 16 + r + 8 * hf;
        ob[((size_t)(b * S_LEN + si)) * DMODEL + h * DK + j * 16 + nn] = (bf16)oacc[t][j][r];
      }
}

// ---------------------------------------------------------------------------
// Final projection: out = attn_out @ Wfc^T + bfc  (fp32 output), 32x64/wave.
__global__ void final_gemm(const bf16* __restrict__ ob, const bf16* __restrict__ wfct,
                           const float* __restrict__ bfc, float* __restrict__ out) {
  int wid   = blockIdx.x * (blockDim.x >> 5) + (threadIdx.x >> 5);
  int mtile = wid >> 4;                // 0..127
  int ntile = wid & 15;
  const bf16* A = ob + (size_t)mtile * 32 * DMODEL;

  v8f acc[2][4] = {};
  for (int k0 = 0; k0 < DMODEL; k0 += 32) {
    v16bf af0 = load_a_frag(A + k0, DMODEL);
    v16bf af1 = load_a_frag(A + (size_t)16 * DMODEL + k0, DMODEL);
#pragma unroll
    for (int j = 0; j < 4; ++j) {
      v16bf bfr = load_b_frag(wfct + (size_t)k0 * DMODEL + ntile * 64 + j * 16, DMODEL);
      acc[0][j] = wmma_bf16(af0, bfr, acc[0][j]);
      acc[1][j] = wmma_bf16(af1, bfr, acc[1][j]);
    }
  }

  int lane = threadIdx.x & 31, nn = lane & 15, hf = lane >> 4;
#pragma unroll
  for (int j = 0; j < 4; ++j) {
    float bias = bfc[ntile * 64 + j * 16 + nn];
#pragma unroll
    for (int t = 0; t < 2; ++t)
#pragma unroll
      for (int r = 0; r < 8; ++r) {
        int mg = mtile * 32 + t * 16 + r + 8 * hf;
        out[(size_t)mg * DMODEL + ntile * 64 + j * 16 + nn] = acc[t][j][r] + bias;
      }
  }
}

// ---------------------------------------------------------------------------
extern "C" void kernel_launch(void* const* d_in, const int* in_sizes, int n_in,
                              void* d_out, int out_size, void* d_ws, size_t ws_size,
                              hipStream_t stream) {
  (void)in_sizes; (void)n_in; (void)out_size;
  const float* x    = (const float*)d_in[0];
  const float* mask = (const float*)d_in[1];
  const float* Wq   = (const float*)d_in[2];
  const float* Wk   = (const float*)d_in[3];
  const float* Wv   = (const float*)d_in[4];
  const float* Wfc  = (const float*)d_in[5];
  const float* bfc  = (const float*)d_in[6];
  float* out = (float*)d_out;

  if (ws_size < (size_t)25165824 * sizeof(bf16)) return;  // need 48 MB staging
  bf16* ws   = (bf16*)d_ws;
  bf16* xb   = ws;                  // [4096,1024]
  bf16* wqt  = ws + 4194304;        // [1024,1024] transposed
  bf16* wkt  = ws + 5242880;
  bf16* wvt  = ws + 6291456;
  bf16* wfct = ws + 7340032;
  bf16* qb   = ws + 8388608;        // [B,H,S,dk]
  bf16* ktb  = ws + 12582912;       // [B,H,dk,S]
  bf16* vb   = ws + 16777216;       // [B,H,S,dk]
  bf16* ob   = ws + 20971520;       // [B,S,D]

  cvt_f32_bf16<<<4194304 / 256, 256, 0, stream>>>(x, xb, 4194304);
  transpose_cvt<<<4096, 256, 0, stream>>>(Wq,  wqt);
  transpose_cvt<<<4096, 256, 0, stream>>>(Wk,  wkt);
  transpose_cvt<<<4096, 256, 0, stream>>>(Wv,  wvt);
  transpose_cvt<<<4096, 256, 0, stream>>>(Wfc, wfct);
  qkv_gemm  <<<768, 256, 0, stream>>>(xb, wqt, wkt, wvt, qb, ktb, vb);
  attn_relu <<<256, 256, 0, stream>>>(qb, ktb, vb, mask, ob);
  final_gemm<<<256, 256, 0, stream>>>(ob, wfct, bfc, out);
}
